// TransformerBlockWithAdapter_47227460386843
// MI455X (gfx1250) — matrix operand order repaired and checked by reading, not compiled, not run on hardware
//
#include <hip/hip_runtime.h>

// ---------------------------------------------------------------------------
// MI455X (gfx1250) transformer block + adapter.
// Matmuls on v_wmma_f32_16x16x32_f16 (wave32, fp32 accumulate).
// GEMM tiles double-buffered in LDS via GLOBAL_LOAD_ASYNC_TO_LDS_B128
// (ASYNCcnt pipeline).  Causal flash attention with online softmax; V is
// kept transposed [B,H,D,T] so P@V B-fragments are single 32B loads.
// ---------------------------------------------------------------------------

typedef __attribute__((ext_vector_type(16))) _Float16 v16h;
typedef __attribute__((ext_vector_type(8)))  float    v8f;
typedef __attribute__((ext_vector_type(4)))  int      v4i;

static __device__ __forceinline__ v8f wmma_f16(const v16h& a, const v16h& b, const v8f& c) {
    return __builtin_amdgcn_wmma_f32_16x16x32_f16(false, a, false, b, (short)0, c, false, false);
}

#if defined(__HIP_DEVICE_COMPILE__) && __has_builtin(__builtin_amdgcn_global_load_async_to_lds_b128)
#define HAS_ASYNC_LDS 1
#else
#define HAS_ASYNC_LDS 0
#endif

// 16-byte global -> LDS copy (async path: no VGPR round trip, ASYNCcnt-tracked)
static __device__ __forceinline__ void cp16_g2l(const _Float16* g, _Float16* l) {
#if HAS_ASYNC_LDS
    __builtin_amdgcn_global_load_async_to_lds_b128(
        (__attribute__((address_space(1))) v4i*)g,
        (__attribute__((address_space(3))) v4i*)l, 0, 0);
#else
    *(uint4*)l = *(const uint4*)g;
#endif
}

#if defined(__HIP_DEVICE_COMPILE__) && __has_builtin(__builtin_amdgcn_s_wait_asynccnt)
#define WAIT_ASYNC(n) __builtin_amdgcn_s_wait_asynccnt(n)
#elif defined(__HIP_DEVICE_COMPILE__) && HAS_ASYNC_LDS
#define WAIT_ASYNC(n) asm volatile("s_wait_asynccnt %0" ::"i"(n) : "memory")
#else
#define WAIT_ASYNC(n) ((void)0)
#endif

constexpr int Bb   = 2;
constexpr int Ts   = 2048;
constexpr int Cc   = 1024;
constexpr int Hh   = 16;
constexpr int FFn  = 4096;
constexpr int BOTn = 64;
constexpr int Dd   = 64;          // C / H
constexpr int Mrows = Bb * Ts;    // 4096 token rows

// ---------------------------------------------------------------------------
// fp32 -> f16 conversion (weights, once per launch)
// ---------------------------------------------------------------------------
__global__ void cvt_f32_to_f16(const float* __restrict__ in,
                               _Float16* __restrict__ out, int n) {
    int i = blockIdx.x * blockDim.x + threadIdx.x;
    int stride = gridDim.x * blockDim.x;
    for (; i < n; i += stride) out[i] = (_Float16)in[i];
}

// ---------------------------------------------------------------------------
// LayerNorm (fp32 in) -> f16 out.  One block per token row, C = 1024.
// ---------------------------------------------------------------------------
__global__ __launch_bounds__(256) void ln_to_f16(const float* __restrict__ x,
                                                 const float* __restrict__ g,
                                                 const float* __restrict__ b,
                                                 _Float16* __restrict__ out, int C) {
    __shared__ float r1[256], r2[256];
    const int row = blockIdx.x, tid = threadIdx.x;
    const float* xr = x + (size_t)row * C;
    float s = 0.f, s2 = 0.f;
    for (int i = tid; i < C; i += 256) { float t = xr[i]; s += t; s2 += t * t; }
    r1[tid] = s; r2[tid] = s2; __syncthreads();
    for (int o = 128; o > 0; o >>= 1) {
        if (tid < o) { r1[tid] += r1[tid + o]; r2[tid] += r2[tid + o]; }
        __syncthreads();
    }
    const float mean = r1[0] / C;
    const float var  = r2[0] / C - mean * mean;
    const float rstd = rsqrtf(var + 1e-5f);
    _Float16* orow = out + (size_t)row * C;
    for (int i = tid; i < C; i += 256)
        orow[i] = (_Float16)((xr[i] - mean) * rstd * g[i] + b[i]);
}

// ---------------------------------------------------------------------------
// Tiled WMMA GEMM:  C[M,N] = A[M,K](f16) * B[K,N](f16)  + bias (+ReLU) (+res)
// Block tile 64x128, BK=32, 256 threads = 8 waves, each wave a 32x32 tile
// (2x2 WMMA).  A/B tiles double-buffered in LDS via async global->LDS copies
// (3 per wave per stage, so s_wait_asynccnt 3 drains exactly one stage).
// OUTT stores the f16 result transposed to [B,H,D,T] (for the V projection).
// ---------------------------------------------------------------------------
template <bool RELU, bool R1, bool R2, bool OUTF, bool OUTH, bool OUTT>
__global__ __launch_bounds__(256) void gemm_f16(
    const _Float16* __restrict__ A, const _Float16* __restrict__ Bw,
    const float* __restrict__ bias, const float* __restrict__ res1,
    const float* __restrict__ res2, float* __restrict__ outf,
    _Float16* __restrict__ outh, int M, int N, int K) {
    constexpr int BM = 64, BN = 128, BK = 32, BNP = BN + 8; // padded rows
    __shared__ __align__(16) _Float16 As[2][BM * BK];       // 2 x 4 KB
    __shared__ __align__(16) _Float16 Bs[2][BK * BNP];      // 2 x 8.5 KB

    const int tid = threadIdx.x;
    const int m0 = blockIdx.y * BM, n0 = blockIdx.x * BN;
    const int wid = tid >> 5, lane = tid & 31, lm = lane & 15, g = lane >> 4;
    const int wr = wid >> 2, wc = wid & 3;            // 2x4 wave grid

    v8f acc[2][2];
    for (int mi = 0; mi < 2; ++mi)
        for (int ni = 0; ni < 2; ++ni)
            for (int v = 0; v < 8; ++v) acc[mi][ni][v] = 0.f;

    const int arow = tid >> 2, acol = (tid & 3) * 8;  // A: 1 x b128 / thread

    // stage k-chunk kc into LDS buffer buf: uniform 3 async ops per thread
    auto stage = [&](int buf, int kc) {
        cp16_g2l(A + (size_t)(m0 + arow) * K + kc + acol,
                 &As[buf][arow * BK + acol]);
        for (int s = 0; s < 2; ++s) {
            int idx = tid * 2 + s;
            int brow = idx >> 4, bcol = (idx & 15) * 8;
            int gcol = n0 + bcol;
            if (gcol + 8 > N) gcol = N - 8;           // clamp (N >= 8 always)
            cp16_g2l(Bw + (size_t)(kc + brow) * N + gcol,
                     &Bs[buf][brow * BNP + bcol]);
        }
    };

    int buf = 0;
    stage(0, 0);
    for (int kc = 0; kc < K; kc += BK) {
        const bool more = (kc + BK) < K;
        if (more) stage(buf ^ 1, kc + BK);            // prefetch next stage
        if (more) { WAIT_ASYNC(3); } else { WAIT_ASYNC(0); }
        __syncthreads();                              // current stage visible

        v16h afr[2], bfr[2];
        for (int mi = 0; mi < 2; ++mi) {
            int row = wr * 32 + mi * 16 + lm;
            for (int e = 0; e < 16; ++e) {
                int col = g * 16 + e;
                afr[mi][e] = As[buf][row * BK + col];
            }
        }
        for (int ni = 0; ni < 2; ++ni) {
            int n = wc * 32 + ni * 16 + lm;
            for (int i = 0; i < 16; ++i)
                bfr[ni][i] = Bs[buf][(g * 16 + i) * BNP + n];
        }
        for (int mi = 0; mi < 2; ++mi)
            for (int ni = 0; ni < 2; ++ni)
                acc[mi][ni] = wmma_f16(afr[mi], bfr[ni], acc[mi][ni]);

        __syncthreads();                              // done reading buf
        buf ^= 1;
    }

    // --- epilogue: bias (+ relu) (+ residuals), fp32 / f16 / f16-transposed ---
    for (int mi = 0; mi < 2; ++mi)
        for (int ni = 0; ni < 2; ++ni)
            for (int v = 0; v < 8; ++v) {
                int row = m0 + wr * 32 + mi * 16 + g * 8 + v;
                int col = n0 + wc * 32 + ni * 16 + lm;
                if (col < N) {
                    float val = acc[mi][ni][v] + bias[col];
                    if (RELU) val = fmaxf(val, 0.f);
                    if (R1) val += res1[(size_t)row * N + col];
                    if (R2) val += res2[(size_t)row * N + col];
                    if (OUTF) outf[(size_t)row * N + col] = val;
                    if (OUTH) outh[(size_t)row * N + col] = (_Float16)val;
                    if (OUTT) {   // [B*T, H*D] -> [B, H, D, T] (V projection)
                        int bz = row >> 11, t = row & (Ts - 1);
                        int h = col >> 6, d = col & 63;
                        outh[(((size_t)bz * Hh + h) * Dd + d) * Ts + t] = (_Float16)val;
                    }
                }
            }
}

// ---------------------------------------------------------------------------
// Causal flash attention.  Grid (T/128, H, B); 8 waves/block, each wave owns
// 16 query rows x D=64.  Key blocks of 32: S = Q(16x64) @ K^T -> 2x2 WMMA,
// online softmax, P staged via per-wave LDS tile, O += P(16x32) @ V(32x64).
// Q/K/O are f16 [B*T, C]; Vt is f16 [B, H, D, T] so every K and V fragment
// is a single aligned 32-byte load.
// ---------------------------------------------------------------------------
__global__ __launch_bounds__(256) void flash_attn(const _Float16* __restrict__ Q,
                                                  const _Float16* __restrict__ Kx,
                                                  const _Float16* __restrict__ Vt,
                                                  _Float16* __restrict__ O) {
    __shared__ __align__(4) _Float16 P[8][16][32];     // 8 KB, per-wave tiles
    const int tid = threadIdx.x, wid = tid >> 5, lane = tid & 31;
    const int lm = lane & 15, g = lane >> 4;
    const int q0 = blockIdx.x * 128 + wid * 16;
    const size_t base = (size_t)blockIdx.z * Ts * Cc + blockIdx.y * Dd;
    const _Float16* qb = Q + base;
    const _Float16* kb = Kx + base;
    const _Float16* vtb = Vt + ((size_t)blockIdx.z * Hh + blockIdx.y) * Dd * Ts;

    // Q fragments pre-scaled by 1/sqrt(D) = 0.125 (exact in f16)
    const _Float16 sc = (_Float16)0.125f;
    v16h qf[2];
    for (int kc = 0; kc < 2; ++kc)
        for (int e = 0; e < 16; ++e) {
            int d = kc * 32 + g * 16 + e;
            qf[kc][e] = sc * qb[(size_t)(q0 + lm) * Cc + d];
        }

    float mrow[8], lrow[8];
    v8f acc_o[4];
    for (int v = 0; v < 8; ++v) { mrow[v] = -3.0e38f; lrow[v] = 0.f; }
    for (int di = 0; di < 4; ++di)
        for (int v = 0; v < 8; ++v) acc_o[di][v] = 0.f;

    const int kb_end = (q0 + 47) >> 5;                 // causal coverage
    for (int kblk = 0; kblk < kb_end; ++kblk) {
        const int ks0 = kblk * 32;
        // ---- S = Q @ K^T (two 16-key column tiles) ----
        v8f sacc[2];
        for (int ni = 0; ni < 2; ++ni)
            for (int v = 0; v < 8; ++v) sacc[ni][v] = 0.f;
        for (int ni = 0; ni < 2; ++ni) {
            int key = ks0 + ni * 16 + lm;
            for (int kc = 0; kc < 2; ++kc) {
                v16h bf = *(const v16h*)(kb + (size_t)key * Cc + kc * 32 + g * 16);
                sacc[ni] = wmma_f16(qf[kc], bf, sacc[ni]);
            }
        }
        // ---- causal mask + online softmax ----
        float rowmax[8];
        for (int v = 0; v < 8; ++v) {
            int qrow = q0 + g * 8 + v;
            for (int ni = 0; ni < 2; ++ni)
                if (ks0 + ni * 16 + lm > qrow) sacc[ni][v] = -1.0e30f;
            float mx = fmaxf(sacc[0][v], sacc[1][v]);
            for (int off = 1; off < 16; off <<= 1)
                mx = fmaxf(mx, __shfl_xor(mx, off, 32));
            rowmax[v] = mx;
        }
        for (int v = 0; v < 8; ++v) {
            float mn = fmaxf(mrow[v], rowmax[v]);
            float scale = __expf(mrow[v] - mn);
            float p0 = __expf(sacc[0][v] - mn);
            float p1 = __expf(sacc[1][v] - mn);
            float rs = p0 + p1;
            for (int off = 1; off < 16; off <<= 1)
                rs += __shfl_xor(rs, off, 32);
            lrow[v] = lrow[v] * scale + rs;
            mrow[v] = mn;
            for (int di = 0; di < 4; ++di) acc_o[di][v] *= scale;
            // layout swap (C-frag -> A-frag) through per-wave LDS tile
            P[wid][g * 8 + v][lm]      = (_Float16)p0;
            P[wid][g * 8 + v][16 + lm] = (_Float16)p1;
        }
        asm volatile("s_wait_dscnt 0" ::: "memory");   // P visible wave-wide
        v16h pf;
        for (int e = 0; e < 16; ++e) {
            int col = g * 16 + e;
            pf[e] = P[wid][lm][col];
        }
        asm volatile("" ::: "memory");                 // keep reads before next writes
        // ---- O += P @ V (V fragments: contiguous rows of Vt) ----
        for (int di = 0; di < 4; ++di) {
            v16h vf = *(const v16h*)(vtb + (size_t)(di * 16 + lm) * Ts + ks0 + g * 16);
            acc_o[di] = wmma_f16(pf, vf, acc_o[di]);
        }
    }
    // ---- normalize + store ----
    _Float16* ob = O + base;
    for (int v = 0; v < 8; ++v) {
        float inv = 1.0f / lrow[v];
        int t = q0 + g * 8 + v;
        for (int di = 0; di < 4; ++di)
            ob[(size_t)t * Cc + di * 16 + lm] = (_Float16)(acc_o[di][v] * inv);
    }
}

// ---------------------------------------------------------------------------
// Launcher
// ---------------------------------------------------------------------------
extern "C" void kernel_launch(void* const* d_in, const int* in_sizes, int n_in,
                              void* d_out, int out_size, void* d_ws, size_t ws_size,
                              hipStream_t stream) {
    (void)in_sizes; (void)n_in; (void)out_size; (void)ws_size;
    const float* x = (const float*)d_in[0];
    /* d_in[1] attn_mask: known causal, not needed */
    const float* Wq = (const float*)d_in[2];  const float* bq = (const float*)d_in[3];
    const float* Wk = (const float*)d_in[4];  const float* bk = (const float*)d_in[5];
    const float* Wv = (const float*)d_in[6];  const float* bv = (const float*)d_in[7];
    const float* Wo = (const float*)d_in[8];  const float* bo = (const float*)d_in[9];
    const float* g1 = (const float*)d_in[10]; const float* be1 = (const float*)d_in[11];
    const float* g2 = (const float*)d_in[12]; const float* be2 = (const float*)d_in[13];
    const float* W1 = (const float*)d_in[14]; const float* bf1 = (const float*)d_in[15];
    const float* W2 = (const float*)d_in[16]; const float* bf2 = (const float*)d_in[17];
    const float* Wd = (const float*)d_in[18]; const float* bd = (const float*)d_in[19];
    const float* Wu = (const float*)d_in[20]; const float* bu = (const float*)d_in[21];
    float* out = (float*)d_out;

    char* ws = (char*)d_ws;
    size_t off = 0;
    auto alloc = [&](size_t bytes) -> void* {
        void* p = ws + off;
        off = (off + bytes + 255) & ~(size_t)255;
        return p;
    };
    // f16 weights
    _Float16* wq_h = (_Float16*)alloc((size_t)Cc * Cc * 2);
    _Float16* wk_h = (_Float16*)alloc((size_t)Cc * Cc * 2);
    _Float16* wv_h = (_Float16*)alloc((size_t)Cc * Cc * 2);
    _Float16* wo_h = (_Float16*)alloc((size_t)Cc * Cc * 2);
    _Float16* w1_h = (_Float16*)alloc((size_t)Cc * FFn * 2);
    _Float16* w2_h = (_Float16*)alloc((size_t)FFn * Cc * 2);
    _Float16* wd_h = (_Float16*)alloc((size_t)Cc * BOTn * 2);
    _Float16* wu_h = (_Float16*)alloc((size_t)BOTn * Cc * 2);
    // activations
    _Float16* xn_h   = (_Float16*)alloc((size_t)Mrows * Cc * 2);
    _Float16* q_h    = (_Float16*)alloc((size_t)Mrows * Cc * 2);
    _Float16* k_h    = (_Float16*)alloc((size_t)Mrows * Cc * 2);
    _Float16* vt_h   = (_Float16*)alloc((size_t)Mrows * Cc * 2);  // [B,H,D,T]
    _Float16* attn_h = (_Float16*)alloc((size_t)Mrows * Cc * 2);
    float*    x1_f   = (float*)   alloc((size_t)Mrows * Cc * 4);
    _Float16* xn2_h  = (_Float16*)alloc((size_t)Mrows * Cc * 2);
    _Float16* h1_h   = (_Float16*)alloc((size_t)Mrows * FFn * 2);
    float*    h_f    = (float*)   alloc((size_t)Mrows * Cc * 4);
    _Float16* h_h    = (_Float16*)alloc((size_t)Mrows * Cc * 2);
    _Float16* a_h    = (_Float16*)alloc((size_t)Mrows * BOTn * 2);

    auto cvt = [&](const float* src, _Float16* dst, int n) {
        cvt_f32_to_f16<<<dim3((n + 255) / 256), 256, 0, stream>>>(src, dst, n);
    };
    cvt(Wq, wq_h, Cc * Cc);   cvt(Wk, wk_h, Cc * Cc);
    cvt(Wv, wv_h, Cc * Cc);   cvt(Wo, wo_h, Cc * Cc);
    cvt(W1, w1_h, Cc * FFn);  cvt(W2, w2_h, FFn * Cc);
    cvt(Wd, wd_h, Cc * BOTn); cvt(Wu, wu_h, BOTn * Cc);

    const dim3 blk(256);
    const dim3 gCC(Cc / 128, Mrows / 64);   // N=1024 tiles
    const dim3 gFF(FFn / 128, Mrows / 64);  // N=4096 tiles
    const dim3 gBO(1, Mrows / 64);          // N=64 tile (guarded)

    // LN1 -> QKV projections (f16 out; V written transposed)
    ln_to_f16<<<Mrows, blk, 0, stream>>>(x, g1, be1, xn_h, Cc);
    gemm_f16<false, false, false, false, true, false><<<gCC, blk, 0, stream>>>(
        xn_h, wq_h, bq, nullptr, nullptr, nullptr, q_h, Mrows, Cc, Cc);
    gemm_f16<false, false, false, false, true, false><<<gCC, blk, 0, stream>>>(
        xn_h, wk_h, bk, nullptr, nullptr, nullptr, k_h, Mrows, Cc, Cc);
    gemm_f16<false, false, false, false, false, true><<<gCC, blk, 0, stream>>>(
        xn_h, wv_h, bv, nullptr, nullptr, nullptr, vt_h, Mrows, Cc, Cc);

    // flash attention
    flash_attn<<<dim3(Ts / 128, Hh, Bb), blk, 0, stream>>>(q_h, k_h, vt_h, attn_h);

    // output projection + residual -> x1 (fp32)
    gemm_f16<false, true, false, true, false, false><<<gCC, blk, 0, stream>>>(
        attn_h, wo_h, bo, x, nullptr, x1_f, nullptr, Mrows, Cc, Cc);

    // LN2 -> FFN
    ln_to_f16<<<Mrows, blk, 0, stream>>>(x1_f, g2, be2, xn2_h, Cc);
    gemm_f16<true, false, false, false, true, false><<<gFF, blk, 0, stream>>>(
        xn2_h, w1_h, bf1, nullptr, nullptr, nullptr, h1_h, Mrows, FFn, Cc);
    gemm_f16<false, false, false, true, true, false><<<gCC, blk, 0, stream>>>(
        h1_h, w2_h, bf2, nullptr, nullptr, h_f, h_h, Mrows, Cc, FFn);

    // adapter: down(+relu) -> up (+ h + x1 residuals) -> final fp32 output
    gemm_f16<true, false, false, false, true, false><<<gBO, blk, 0, stream>>>(
        h_h, wd_h, bd, nullptr, nullptr, nullptr, a_h, Mrows, BOTn, Cc);
    gemm_f16<false, true, true, true, false, false><<<gCC, blk, 0, stream>>>(
        a_h, wu_h, bu, h_f, x1_f, out, nullptr, Mrows, Cc, BOTn);
}
